// GAGNN_Model_37984690765934
// MI455X (gfx1250) — matrix operand order repaired and checked by reading, not compiled, not run on hardware
//
#include <hip/hip_runtime.h>
#include <hip/hip_fp16.h>

typedef _Float16 f16;
typedef __attribute__((ext_vector_type(16))) _Float16 v16h;
typedef __attribute__((ext_vector_type(8)))  float    v8f;

#define N_NODES 50000
#define N_EDGES 200000
#define EF      (N_EDGES + N_NODES)   // 250000 (edges + self loops)
#define IN_C    128
#define EDGE_C  16
#define HIDC    64
#define HEADS   8
#define OUTD    512
#define MLP_IN  1040
#define KPAD    1056                  // 1040 padded to multiple of 32
#define MTILES  (N_NODES/16)          // 3125
#define ETILES  (N_EDGES/16)          // 12500

// ---------- helpers ----------
__device__ inline float leaky02(float x) { return x >= 0.f ? x : 0.2f * x; }

__device__ inline void atomicMaxF(float* addr, float v) {
  if (v >= 0.f) atomicMax((int*)addr, __float_as_int(v));
  else          atomicMin((unsigned int*)addr, __float_as_uint(v));
}

// ---------- fills ----------
__global__ void k_fill(unsigned int* p, long long n, unsigned int v) {
  long long i = (long long)blockIdx.x * blockDim.x + threadIdx.x;
  long long s = (long long)gridDim.x * blockDim.x;
  for (; i < n; i += s) p[i] = v;
}

// ---------- weight prep ----------
// W_lin [128][512] f32 -> WlinT [512][128] f16
__global__ void k_wlin_t(const float* __restrict__ W, f16* __restrict__ Wt) {
  int i = blockIdx.x * blockDim.x + threadIdx.x;
  if (i >= OUTD * IN_C) return;
  int n = i >> 7, k = i & 127;
  Wt[i] = (f16)W[k * OUTD + n];
}
// W1 [1040][64] f32 -> W1t [64][1056] f16, zero padded
__global__ void k_w1_t(const float* __restrict__ W, f16* __restrict__ Wt) {
  int i = blockIdx.x * blockDim.x + threadIdx.x;
  if (i >= 64 * KPAD) return;
  int n = i / KPAD, k = i % KPAD;
  Wt[i] = (k < MLP_IN) ? (f16)W[k * 64 + n] : (f16)0.f;
}
// Wea[k][h] = sum_c W_edge[k, h*64+c] * att_edge[h,c]   (16x8)
__global__ void k_wea(const float* __restrict__ W_edge, const float* __restrict__ att_edge,
                      float* __restrict__ Wea) {
  int t = threadIdx.x;
  if (t >= 128) return;
  int k = t >> 3, h = t & 7;
  float s = 0.f;
  for (int c = 0; c < HIDC; ++c) s += W_edge[k * OUTD + h * HIDC + c] * att_edge[h * HIDC + c];
  Wea[k * 8 + h] = s;
}
// edge_attr column sums (for mean)
__global__ void k_easum(const float* __restrict__ ea, float* __restrict__ easum) {
  __shared__ float sh[256];
  int tid = threadIdx.x;
  int k = tid & 15, g = tid >> 4;
  float s = 0.f;
  long long base = (long long)blockIdx.x * 1024 + (long long)g * 64;
  for (int i = 0; i < 64; ++i) {
    long long r = base + i;
    if (r < N_EDGES) s += ea[r * EDGE_C + k];
  }
  sh[tid] = s;
  __syncthreads();
  if (tid < 16) {
    float a = 0.f;
    for (int g2 = 0; g2 < 16; ++g2) a += sh[g2 * 16 + tid];
    atomicAdd(&easum[tid], a);
  }
}
// a_edge for self loops (same for every node)
__global__ void k_loopedge(const float* __restrict__ easum, const float* __restrict__ Wea,
                           float* __restrict__ aloop) {
  int h = threadIdx.x;
  if (h >= 8) return;
  const float invE = 1.0f / (float)N_EDGES;
  float s = 0.f;
  for (int k = 0; k < EDGE_C; ++k) s += easum[k] * invE * Wea[k * 8 + h];
  aloop[h] = s;
}

// ---------- GEMM 1: xs = x @ W_lin  (f16 WMMA, f32 accum, store f16) ----------
__global__ void __launch_bounds__(256) k_xw(const float* __restrict__ x,
                                            const f16* __restrict__ WlinT,
                                            f16* __restrict__ xs) {
  int wid  = (blockIdx.x * blockDim.x + threadIdx.x) >> 5;
  int lane = threadIdx.x & 31;
  int mtile = wid >> 3;       // 16-row tile
  int ng    = wid & 7;        // 64-col group
  if (mtile >= MTILES) return;
  int m = lane & 15, hi = lane >> 4;
  const float* xr = x + ((long long)mtile * 16 + m) * IN_C;
  v8f acc[4] = {v8f{}, v8f{}, v8f{}, v8f{}};
  #pragma unroll
  for (int k0 = 0; k0 < IN_C; k0 += 32) {
    v16h a;
    const float* p1 = xr + k0 + hi * 16;
    #pragma unroll
    for (int e = 0; e < 8; ++e) { a[e] = (f16)p1[e]; a[8 + e] = (f16)p1[8 + e]; }
    #pragma unroll
    for (int nt = 0; nt < 4; ++nt) {
      const v16h b = *(const v16h*)(WlinT + (ng * 64 + nt * 16 + (lane & 15)) * IN_C + k0 + hi * 16);
      acc[nt] = __builtin_amdgcn_wmma_f32_16x16x32_f16(false, a, false, b, (short)0, acc[nt], false, false);
    }
  }
  #pragma unroll
  for (int nt = 0; nt < 4; ++nt) {
    #pragma unroll
    for (int r = 0; r < 8; ++r) {
      long long ro = (long long)mtile * 16 + hi * 8 + r;
      xs[ro * OUTD + ng * 64 + nt * 16 + (lane & 15)] = (f16)acc[nt][r];
    }
  }
}

// ---------- per-node attention scores ----------
__global__ void k_attn_node(const f16* __restrict__ xs, const float* __restrict__ att_src,
                            const float* __restrict__ att_dst,
                            float* __restrict__ a_src, float* __restrict__ a_dst) {
  int i = blockIdx.x * blockDim.x + threadIdx.x;
  if (i >= N_NODES * HEADS) return;
  int n = i >> 3, h = i & 7;
  const f16* p = xs + (long long)n * OUTD + h * HIDC;
  float s = 0.f, d = 0.f;
  for (int c = 0; c < HIDC; ++c) {
    float v = (float)p[c];
    s += v * att_src[h * HIDC + c];
    d += v * att_dst[h * HIDC + c];
  }
  a_src[i] = s;
  a_dst[i] = d;
}

// ---------- alpha + segment max ----------
__global__ void k_alpha(const int* __restrict__ rowp, const int* __restrict__ colp,
                        const float* __restrict__ ea, const float* __restrict__ Wea,
                        const float* __restrict__ aloop,
                        const float* __restrict__ a_src, const float* __restrict__ a_dst,
                        float* __restrict__ alpha, float* __restrict__ amax) {
  int e = blockIdx.x * blockDim.x + threadIdx.x;
  if (e >= EF) return;
  int r, c;
  float ae[8];
  if (e < N_EDGES) {
    r = rowp[e]; c = colp[e];
    float eav[16];
    #pragma unroll
    for (int k = 0; k < EDGE_C; ++k) eav[k] = ea[(long long)e * EDGE_C + k];
    #pragma unroll
    for (int h = 0; h < 8; ++h) {
      float s = 0.f;
      #pragma unroll
      for (int k = 0; k < EDGE_C; ++k) s += eav[k] * Wea[k * 8 + h];
      ae[h] = s;
    }
  } else {
    r = c = e - N_EDGES;
    #pragma unroll
    for (int h = 0; h < 8; ++h) ae[h] = aloop[h];
  }
  #pragma unroll
  for (int h = 0; h < 8; ++h) {
    float al = leaky02(a_src[r * 8 + h] + a_dst[c * 8 + h] + ae[h]);
    alpha[(long long)e * 8 + h] = al;
    atomicMaxF(&amax[c * 8 + h], al);
  }
}

// ---------- exp + denom ----------
__global__ void k_expdenom(const int* __restrict__ colp, float* __restrict__ alpha,
                           const float* __restrict__ amax, float* __restrict__ denom) {
  int e = blockIdx.x * blockDim.x + threadIdx.x;
  if (e >= EF) return;
  int c = e < N_EDGES ? colp[e] : e - N_EDGES;
  #pragma unroll
  for (int h = 0; h < 8; ++h) {
    float ex = expf(alpha[(long long)e * 8 + h] - amax[c * 8 + h]);
    alpha[(long long)e * 8 + h] = ex;
    atomicAdd(&denom[c * 8 + h], ex);
  }
}

// ---------- message scatter ----------
__global__ void k_msg(const int* __restrict__ rowp, const int* __restrict__ colp,
                      const f16* __restrict__ xs, const float* __restrict__ expa,
                      const float* __restrict__ denom, float* __restrict__ outacc) {
  long long t = (long long)blockIdx.x * blockDim.x + threadIdx.x;
  if (t >= (long long)EF * 128) return;
  int e  = (int)(t >> 7);
  int c4 = (int)(t & 127);
  int h  = c4 >> 4;
  int r = e < N_EDGES ? rowp[e] : e - N_EDGES;
  int c = e < N_EDGES ? colp[e] : e - N_EDGES;
  float attn = expa[(long long)e * 8 + h] / (denom[c * 8 + h] + 1e-16f);
  const f16* xp = xs + (long long)r * OUTD + c4 * 4;
  float* op = outacc + (long long)c * OUTD + c4 * 4;
  #pragma unroll
  for (int i = 0; i < 4; ++i) atomicAdd(op + i, attn * (float)xp[i]);
}

// ---------- finish: h = leaky(out+bias), store f16, pooled sum ----------
__global__ void __launch_bounds__(512) k_finish(const float* __restrict__ outacc,
                                                const float* __restrict__ bias,
                                                f16* __restrict__ hfeat,
                                                float* __restrict__ zsum) {
  int c = threadIdx.x;
  float b = bias[c];
  float acc = 0.f;
  long long base = (long long)blockIdx.x * 64;
  for (int i = 0; i < 64; ++i) {
    long long n = base + i;
    if (n < N_NODES) {
      float v = leaky02(outacc[n * OUTD + c] + b);
      hfeat[n * OUTD + c] = (f16)v;
      acc += v;
    }
  }
  atomicAdd(&zsum[c], acc);
}

// ---------- group scorer ----------
__global__ void k_group(const float* __restrict__ zsum, const float* __restrict__ Wg1,
                        const float* __restrict__ bg1, const float* __restrict__ Wg2,
                        const float* __restrict__ bg2, float* __restrict__ out) {
  __shared__ float t[32];
  int j = threadIdx.x;
  const float invN = 1.0f / (float)N_NODES;
  if (j < 32) {
    float s = bg1[j];
    for (int c = 0; c < OUTD; ++c) s += zsum[c] * invN * Wg1[c * 32 + j];
    t[j] = s > 0.f ? s : 0.f;
  }
  __syncthreads();
  if (j == 0) {
    float g = bg2[0];
    for (int jj = 0; jj < 32; ++jj) g += t[jj] * Wg2[jj];
    out[2 * N_EDGES] = 1.0f / (1.0f + expf(-g));
  }
}

// ---------- GEMM 2: edge MLP (WMMA) ----------
__device__ inline void load8(v16h* a, int off, int kb, const f16* hr, const f16* hc,
                             const float* eap) {
  if (kb < 512) {
    const f16* p = hr + kb;
    #pragma unroll
    for (int i = 0; i < 8; ++i) (*a)[off + i] = p[i];
  } else if (kb < 1024) {
    const f16* p = hc + (kb - 512);
    #pragma unroll
    for (int i = 0; i < 8; ++i) (*a)[off + i] = p[i];
  } else if (kb < MLP_IN) {
    const float* p = eap + (kb - 1024);
    #pragma unroll
    for (int i = 0; i < 8; ++i) (*a)[off + i] = (f16)p[i];
  } else {
    #pragma unroll
    for (int i = 0; i < 8; ++i) (*a)[off + i] = (f16)0.f;
  }
}

__global__ void __launch_bounds__(256) k_edge_mlp(const f16* __restrict__ hfeat,
        const float* __restrict__ ea, const int* __restrict__ rowp, const int* __restrict__ colp,
        const f16* __restrict__ W1t, const float* __restrict__ b1,
        const float* __restrict__ W2, const float* __restrict__ b2,
        float* __restrict__ out) {
  __shared__ float hid[8][16][64];
  int w = threadIdx.x >> 5, lane = threadIdx.x & 31;
  int tile = blockIdx.x * 8 + w;
  if (tile > ETILES - 1) tile = ETILES - 1;   // keep all lanes/waves live for WMMA + barrier
  int m = lane & 15, hi = lane >> 4;
  int e = tile * 16 + m;
  const f16* hr = hfeat + (long long)rowp[e] * OUTD;
  const f16* hc = hfeat + (long long)colp[e] * OUTD;
  const float* eap = ea + (long long)e * EDGE_C;
  v8f acc[4] = {v8f{}, v8f{}, v8f{}, v8f{}};
  for (int k0 = 0; k0 < KPAD; k0 += 32) {
    v16h a;
    load8(&a, 0, k0 + hi * 16, hr, hc, eap);
    load8(&a, 8, k0 + hi * 16 + 8, hr, hc, eap);
    #pragma unroll
    for (int nt = 0; nt < 4; ++nt) {
      const v16h b = *(const v16h*)(W1t + (nt * 16 + (lane & 15)) * KPAD + k0 + hi * 16);
      acc[nt] = __builtin_amdgcn_wmma_f32_16x16x32_f16(false, a, false, b, (short)0, acc[nt], false, false);
    }
  }
  #pragma unroll
  for (int nt = 0; nt < 4; ++nt) {
    int n = nt * 16 + (lane & 15);
    float bb = b1[n];
    #pragma unroll
    for (int r = 0; r < 8; ++r) {
      float v = acc[nt][r] + bb;
      hid[w][r + hi * 8][n] = v > 0.f ? v : 0.f;
    }
  }
  __syncthreads();
  int m2 = lane >> 1, j = lane & 1;
  float g = b2[j];
  for (int n = 0; n < 64; ++n) g += hid[w][m2][n] * W2[n * 2 + j];
  out[(long long)(tile * 16 + m2) * 2 + j] = g;
}

// ---------- launch ----------
extern "C" void kernel_launch(void* const* d_in, const int* in_sizes, int n_in,
                              void* d_out, int out_size, void* d_ws, size_t ws_size,
                              hipStream_t stream) {
  const float* x        = (const float*)d_in[0];
  const int*   eidx     = (const int*)d_in[1];
  const float* ea       = (const float*)d_in[2];
  const float* W_lin    = (const float*)d_in[3];
  const float* att_src  = (const float*)d_in[4];
  const float* att_dst  = (const float*)d_in[5];
  const float* W_edge   = (const float*)d_in[6];
  const float* att_edge = (const float*)d_in[7];
  const float* gat_bias = (const float*)d_in[8];
  const float* W1       = (const float*)d_in[9];
  const float* b1       = (const float*)d_in[10];
  const float* W2       = (const float*)d_in[11];
  const float* b2       = (const float*)d_in[12];
  const float* Wg1      = (const float*)d_in[13];
  const float* bg1      = (const float*)d_in[14];
  const float* Wg2      = (const float*)d_in[15];
  const float* bg2      = (const float*)d_in[16];
  float* out = (float*)d_out;
  const int* rowp = eidx;            // edge_index[0]
  const int* colp = eidx + N_EDGES;  // edge_index[1]

  // workspace layout (256B aligned slabs)
  char* base = (char*)d_ws;
  size_t o = 0;
  auto alloc = [&](size_t bytes) { size_t r = o; o = (o + bytes + 255) & ~(size_t)255; return r; };
  f16*   xs     = (f16*)  (base + alloc((size_t)N_NODES * OUTD * 2)); // reused as h (f16) later
  float* outacc = (float*)(base + alloc((size_t)N_NODES * OUTD * 4));
  float* alpha  = (float*)(base + alloc((size_t)EF * 8 * 4));
  float* a_src  = (float*)(base + alloc((size_t)N_NODES * 8 * 4));
  float* a_dst  = (float*)(base + alloc((size_t)N_NODES * 8 * 4));
  float* amax   = (float*)(base + alloc((size_t)N_NODES * 8 * 4));
  float* denom  = (float*)(base + alloc((size_t)N_NODES * 8 * 4));
  f16*   WlinT  = (f16*)  (base + alloc((size_t)OUTD * IN_C * 2));
  f16*   W1t    = (f16*)  (base + alloc((size_t)64 * KPAD * 2));
  float* easum  = (float*)(base + alloc(16 * 4));
  float* Wea    = (float*)(base + alloc(128 * 4));
  float* aloop  = (float*)(base + alloc(8 * 4));
  float* zsum   = (float*)(base + alloc(512 * 4));
  f16*   hfeat  = xs;  // h (f16) overwrites xs after message pass

  // init accumulators
  k_fill<<<1024, 256, 0, stream>>>((unsigned int*)outacc, (long long)N_NODES * OUTD, 0u);
  k_fill<<<64,   256, 0, stream>>>((unsigned int*)denom, (long long)N_NODES * 8, 0u);
  k_fill<<<64,   256, 0, stream>>>((unsigned int*)amax,  (long long)N_NODES * 8, 0xFF800000u);
  k_fill<<<1,     32, 0, stream>>>((unsigned int*)easum, 16, 0u);
  k_fill<<<2,    256, 0, stream>>>((unsigned int*)zsum, 512, 0u);

  // weight prep
  k_wlin_t<<<(OUTD * IN_C + 255) / 256, 256, 0, stream>>>(W_lin, WlinT);
  k_w1_t<<<(64 * KPAD + 255) / 256, 256, 0, stream>>>(W1, W1t);
  k_wea<<<1, 128, 0, stream>>>(W_edge, att_edge, Wea);
  k_easum<<<(N_EDGES + 1023) / 1024, 256, 0, stream>>>(ea, easum);
  k_loopedge<<<1, 32, 0, stream>>>(easum, Wea, aloop);

  // GAT
  k_xw<<<MTILES, 256, 0, stream>>>(x, WlinT, xs);
  k_attn_node<<<(N_NODES * HEADS + 255) / 256, 256, 0, stream>>>(xs, att_src, att_dst, a_src, a_dst);
  k_alpha<<<(EF + 255) / 256, 256, 0, stream>>>(rowp, colp, ea, Wea, aloop, a_src, a_dst, alpha, amax);
  k_expdenom<<<(EF + 255) / 256, 256, 0, stream>>>(colp, alpha, amax, denom);
  k_msg<<<(unsigned)(((long long)EF * 128) / 256), 256, 0, stream>>>(rowp, colp, xs, alpha, denom, outacc);
  k_finish<<<(N_NODES + 63) / 64, 512, 0, stream>>>(outacc, gat_bias, hfeat, zsum);

  // heads
  k_group<<<1, 32, 0, stream>>>(zsum, Wg1, bg1, Wg2, bg2, out);
  k_edge_mlp<<<(ETILES + 7) / 8, 256, 0, stream>>>(hfeat, ea, rowp, colp, W1t, b1, W2, b2, out);
}
